// AGTLayer_22522808500915
// MI455X (gfx1250) — matrix operand order repaired and checked by reading, not compiled, not run on hardware
//
#include <hip/hip_runtime.h>
#include <hip/hip_bf16.h>

// ---------------------------------------------------------------------------
// AGT layer for MI455X (gfx1250, wave32, WMMA 16x16x32 bf16).
// Compute-bound (AI ~220 flop/byte) -> all GEMMs + per-node attention run on
// v_wmma_f32_16x16x32_bf16. GEMM tiles stream into LDS via the CDNA5 async
// DMA path (global_load_async_to_lds_b128, ASYNCcnt) with double buffering.
// Fragment layouts per CDNA5 ISA 7.12.2:
//   A (16x32 bf16): lane = m%16 + 16*half; VGPR p holds K pair
//                   k = ((2p&8)<<1) + (2p&7) + 8*half   (pairs contiguous)
//   B (32x16 bf16): lane = n%16 + 16*half; VGPR p holds K pair k = ((2p&8)<<1) + (2p&7) + 8*half
//   C/D (16x16 f32): VGPR r holds (m = r + 8*half, n = lane%16)
// ---------------------------------------------------------------------------

typedef __attribute__((ext_vector_type(16))) __bf16 v16bf;
typedef __attribute__((ext_vector_type(8)))  float  v8f;

union ABFrag { v16bf v; unsigned u[8]; unsigned short s[16]; };

__device__ __forceinline__ unsigned short f2bf(float f) {
    union { float f; unsigned u; } a; a.f = f;
    unsigned u = a.u;
    u += 0x7fffu + ((u >> 16) & 1u);          // round-to-nearest-even
    return (unsigned short)(u >> 16);
}

// Async DMA: LDS[lds_off] = MEM[gaddr], 16 bytes per lane, tracked by ASYNCcnt.
__device__ __forceinline__ void async_ld_b128(unsigned lds_off, const void* gaddr) {
    asm volatile("global_load_async_to_lds_b128 %0, %1, off"
                 :: "v"(lds_off), "v"(gaddr) : "memory");
}
#define WAIT_ASYNC(n) asm volatile("s_wait_asynccnt " #n ::: "memory")

#define NROWS 32768
#define DIM   512

// -------------------------- fp32 -> bf16 convert ---------------------------
__global__ __launch_bounds__(256)
void cvt_bf16_kernel(const float* __restrict__ src, unsigned short* __restrict__ dst) {
    size_t i = (size_t)blockIdx.x * 256 + threadIdx.x;
    dst[i] = f2bf(src[i]);
}

// fp32 W[k][n] -> bf16 WT[n][k] (so GEMM B-tiles are contiguous byte copies)
__global__ __launch_bounds__(256)
void cvt_bf16_t_kernel(const float* __restrict__ W, unsigned short* __restrict__ WT) {
    size_t i = (size_t)blockIdx.x * 256 + threadIdx.x;
    const size_t k = i >> 9, n = i & 511;
    WT[n * DIM + k] = f2bf(W[i]);
}

// ------------------------------ WMMA GEMM ----------------------------------
// C[M x 512] = A[M x 512] (bf16) @ W (given as WT[n][k] bf16), fp32 out.
// 256 thr = 8 waves; tile 128(M) x 64(N); K in steps of 32, double-buffered
// LDS fed by async-to-LDS DMA (3 x b128 per thread per K step).
template <int RELU>
__global__ __launch_bounds__(256)
void gemm_bf16_kernel(const unsigned short* __restrict__ A,
                      const unsigned short* __restrict__ BT,
                      float* __restrict__ C)
{
    __shared__ alignas(16) unsigned short As[2][128 * 32];  // [row][k] 2 x 8 KB
    __shared__ alignas(16) unsigned short Bt[2][64 * 32];   // [col][k] 2 x 4 KB

    const int tid  = threadIdx.x;
    const int wave = tid >> 5, lane = tid & 31;
    const int half = lane >> 4, ml = lane & 15;
    const int m0 = blockIdx.x * 128, n0 = blockIdx.y * 64;

    // staging coordinates (pure contiguous copies)
    const int ar = tid >> 1, akp = (tid & 1) << 4;   // A: 128 rows x 32 k
    const int br = tid >> 2, bkp = (tid & 3) << 3;   // B: 64 cols x 32 k
    const unsigned short* aSrc = A  + (size_t)(m0 + ar) * DIM + akp;
    const unsigned short* bSrc = BT + (size_t)(n0 + br) * DIM + bkp;
    const unsigned aOff = (unsigned)(size_t)&As[0][ar * 32 + akp];
    const unsigned bOff = (unsigned)(size_t)&Bt[0][br * 32 + bkp];
    const unsigned aStr = (unsigned)((size_t)&As[1][0] - (size_t)&As[0][0]);
    const unsigned bStr = (unsigned)((size_t)&Bt[1][0] - (size_t)&Bt[0][0]);

    auto stage = [&](int buf, int kk) {
        async_ld_b128(aOff + buf * aStr,      aSrc + kk);
        async_ld_b128(aOff + buf * aStr + 16, aSrc + kk + 8);
        async_ld_b128(bOff + buf * bStr,      bSrc + kk);
    };

    v8f acc[4] = {};
    stage(0, 0);

    for (int kk = 0; kk < DIM; kk += 32) {
        const int buf = (kk >> 5) & 1;
        if (kk + 32 < DIM) {
            stage(buf ^ 1, kk + 32);     // prefetch next tile into other buffer
            WAIT_ASYNC(0x3);             // in-order: first 3 (current buf) done
        } else {
            WAIT_ASYNC(0x0);
        }
        __syncthreads();

        const unsigned short* as = &As[buf][0];
        const unsigned short* bt = &Bt[buf][0];

        ABFrag af;
        #pragma unroll
        for (int p = 0; p < 8; ++p) {
            const int j = 2 * p;
            const int k = ((j & 8) << 1) + (j & 7) + half * 8;
            af.u[p] = *(const unsigned*)(as + (wave * 16 + ml) * 32 + k);
        }
        #pragma unroll
        for (int nt = 0; nt < 4; ++nt) {
            ABFrag bfg;
            #pragma unroll
            for (int p = 0; p < 8; ++p)
                bfg.u[p] = *(const unsigned*)(bt + (nt * 16 + ml) * 32 + (((2 * p & 8) << 1) + (2 * p & 7) + 8 * half));
            acc[nt] = __builtin_amdgcn_wmma_f32_16x16x32_bf16(
                false, af.v, false, bfg.v, (short)0, acc[nt], false, false);
        }
        __syncthreads();                 // all waves done reading buf
    }

    #pragma unroll
    for (int nt = 0; nt < 4; ++nt) {
        const int col = n0 + nt * 16 + ml;
        #pragma unroll
        for (int r = 0; r < 8; ++r) {
            const int row = m0 + wave * 16 + r + 8 * half;
            float vv = acc[nt][r];
            if (RELU) vv = vv > 0.f ? vv : 0.f;
            C[(size_t)row * DIM + col] = vv;
        }
    }
}

// ----------------------- per-node linear attention -------------------------
// One wave per node. kv = K^T V (64x8x64), num = Q kv (16x64x64), both WMMA.
__global__ __launch_bounds__(128)
void attn_kernel(const float* __restrict__ q, const float* __restrict__ k,
                 const float* __restrict__ v, unsigned short* __restrict__ attnb)
{
    __shared__ alignas(16) unsigned short kvs[4][64 * 64]; // [wave][e*64+d] 32 KB
    __shared__ float ksumS[4][64];
    __shared__ float denomS[4][8];

    const int wave = threadIdx.x >> 5, lane = threadIdx.x & 31;
    const int half = lane >> 4, ml = lane & 15;
    const size_t b = (size_t)blockIdx.x * 4 + wave;
    const float* qb = q + b * DIM;
    const float* kb = k + b * DIM;
    const float* vb = v + b * DIM;

    // --- kv = K^T V : A = K^T (64x8 padded to 16x32 tiles), B = V (8x64 padded)
    ABFrag af[4], bf[4];
    #pragma unroll
    for (int t = 0; t < 4; ++t) {
        #pragma unroll
        for (int p = 0; p < 8; ++p) { af[t].u[p] = 0; bf[t].u[p] = 0; }
        if (half == 0) {
            #pragma unroll
            for (int j = 0; j < 8; ++j) {        // element j -> head n=j (k index)
                af[t].s[j] = f2bf(kb[j * 64 + t * 16 + ml]);   // K[n][d], m=d
                bf[t].s[j] = f2bf(vb[j * 64 + t * 16 + ml]);   // V[n][e], n=e col
            }
        }
    }
    v8f kv[4][4];
    #pragma unroll
    for (int dt = 0; dt < 4; ++dt)
        #pragma unroll
        for (int et = 0; et < 4; ++et) {
            v8f z = {};
            kv[dt][et] = __builtin_amdgcn_wmma_f32_16x16x32_bf16(
                false, af[dt].v, false, bf[et].v, (short)0, z, false, false);
        }

    // --- ksum[d] = sum_n K[n][d]
    {
        float s0 = 0.f, s1 = 0.f;
        #pragma unroll
        for (int n = 0; n < 8; ++n) { s0 += kb[n * 64 + lane]; s1 += kb[n * 64 + 32 + lane]; }
        ksumS[wave][lane] = s0; ksumS[wave][lane + 32] = s1;
    }

    // --- spill kv to LDS, e-major so num's B fragments read contiguous k-pairs
    #pragma unroll
    for (int dt = 0; dt < 4; ++dt)
        #pragma unroll
        for (int et = 0; et < 4; ++et)
            #pragma unroll
            for (int r = 0; r < 8; ++r) {
                const int d = dt * 16 + r + 8 * half;
                const int e = et * 16 + ml;
                kvs[wave][e * 64 + d] = f2bf(kv[dt][et][r]);
            }
    __syncthreads();

    // --- denom[n] = q[n,:] . ksum
    if (lane < 8) {
        float ds = 0.f;
        #pragma unroll 8
        for (int d = 0; d < 64; ++d) ds += qb[lane * 64 + d] * ksumS[wave][d];
        denomS[wave][lane] = ds;
    }

    // --- num = Q @ kv : M=16 (8 valid heads), K=64 (2 steps), N=64 (4 tiles)
    v8f num[4] = {};
    #pragma unroll
    for (int kt = 0; kt < 2; ++kt) {
        ABFrag aq;
        #pragma unroll
        for (int p = 0; p < 8; ++p) aq.u[p] = 0;
        #pragma unroll
        for (int j = 0; j < 16; ++j) {
            const int d = kt * 32 + ((j & 8) << 1) + (j & 7) + 8 * half;
            aq.s[j] = (ml < 8) ? f2bf(qb[ml * 64 + d]) : (unsigned short)0;
        }
        #pragma unroll
        for (int et = 0; et < 4; ++et) {
            ABFrag bq;
            #pragma unroll
            for (int p = 0; p < 8; ++p)
                bq.u[p] = *(const unsigned*)(&kvs[wave][(et * 16 + ml) * 64 + kt * 32 + ((2 * p & 8) << 1) + (2 * p & 7) + 8 * half]);
            num[et] = __builtin_amdgcn_wmma_f32_16x16x32_bf16(
                false, aq.v, false, bq.v, (short)0, num[et], false, false);
        }
    }
    __syncthreads();

    // --- attn = num / denom, stored bf16 for the Wf GEMM
    #pragma unroll
    for (int et = 0; et < 4; ++et)
        #pragma unroll
        for (int r = 0; r < 8; ++r) {
            const int m = r + 8 * half;         // head
            if (m < 8) {
                const float vv = num[et][r] / denomS[wave][m];
                attnb[b * DIM + m * 64 + et * 16 + ml] = f2bf(vv);
            }
        }
}

// ------------------------------ LayerNorm ----------------------------------
// out = LN(h + y), one wave per row, wave32 shuffle reduction.
__global__ __launch_bounds__(256)
void ln_kernel(const float* __restrict__ h, const float* __restrict__ y,
               const float* __restrict__ g, const float* __restrict__ be,
               float* __restrict__ out)
{
    const int wave = threadIdx.x >> 5, lane = threadIdx.x & 31;
    const size_t row = (size_t)blockIdx.x * 8 + wave;
    const float* hr = h + row * DIM;
    const float* yr = y + row * DIM;

    float x[16], s = 0.f, ss = 0.f;
    #pragma unroll
    for (int t = 0; t < 16; ++t) {
        const int i = lane + t * 32;
        const float xv = hr[i] + yr[i];
        x[t] = xv; s += xv; ss += xv * xv;
    }
    #pragma unroll
    for (int o = 16; o > 0; o >>= 1) {
        s  += __shfl_xor(s,  o, 32);
        ss += __shfl_xor(ss, o, 32);
    }
    const float mu   = s * (1.f / 512.f);
    const float var  = ss * (1.f / 512.f) - mu * mu;
    const float rstd = rsqrtf(var + 1e-5f);
    float* orow = out + row * DIM;
    #pragma unroll
    for (int t = 0; t < 16; ++t) {
        const int i = lane + t * 32;
        orow[i] = (x[t] - mu) * rstd * g[i] + be[i];
    }
}

// ------------------------------ launcher -----------------------------------
extern "C" void kernel_launch(void* const* d_in, const int* in_sizes, int n_in,
                              void* d_out, int out_size, void* d_ws, size_t ws_size,
                              hipStream_t stream)
{
    const float* h  = (const float*)d_in[0];
    const float* Wq = (const float*)d_in[1];
    const float* Wk = (const float*)d_in[2];
    const float* Wv = (const float*)d_in[3];
    const float* Wf = (const float*)d_in[4];
    const float* g  = (const float*)d_in[5];
    const float* be = (const float*)d_in[6];

    const size_t N = NROWS, D = DIM, ND = N * D;
    float* outLN = (float*)d_out;
    float* qo = outLN + ND;
    float* ko = outLN + 2 * ND;
    float* vo = outLN + 3 * ND;

    char* ws = (char*)d_ws;
    unsigned short* hb    = (unsigned short*)(ws);                 // 32 MB
    unsigned short* attnb = (unsigned short*)(ws + ND * 2);        // 32 MB
    float*          y     = (float*)(ws + ND * 4);                 // 64 MB
    unsigned short* wqb   = (unsigned short*)(ws + ND * 8);        // 4 x 0.5 MB (transposed)
    unsigned short* wkb   = wqb + D * D;
    unsigned short* wvb   = wkb + D * D;
    unsigned short* wfb   = wvb + D * D;

    cvt_bf16_kernel  <<<ND / 256,    256, 0, stream>>>(h,  hb);
    cvt_bf16_t_kernel<<<D * D / 256, 256, 0, stream>>>(Wq, wqb);
    cvt_bf16_t_kernel<<<D * D / 256, 256, 0, stream>>>(Wk, wkb);
    cvt_bf16_t_kernel<<<D * D / 256, 256, 0, stream>>>(Wv, wvb);
    cvt_bf16_t_kernel<<<D * D / 256, 256, 0, stream>>>(Wf, wfb);

    dim3 gg(N / 128, D / 64);
    gemm_bf16_kernel<1><<<gg, 256, 0, stream>>>(hb, wqb, qo);
    gemm_bf16_kernel<1><<<gg, 256, 0, stream>>>(hb, wkb, ko);
    gemm_bf16_kernel<0><<<gg, 256, 0, stream>>>(hb, wvb, vo);

    attn_kernel<<<N / 4, 128, 0, stream>>>(qo, ko, vo, attnb);

    gemm_bf16_kernel<0><<<gg, 256, 0, stream>>>(attnb, wfb, y);

    ln_kernel<<<N / 8, 256, 0, stream>>>(h, y, g, be, outLN);
}
